// EncoderBlock_69767448756345
// MI455X (gfx1250) — matrix operand order repaired and checked by reading, not compiled, not run on hardware
//
#include <hip/hip_runtime.h>
#include <hip/hip_bf16.h>
#include <stdint.h>

// ---------------------------------------------------------------------------
// Problem dims (compile-time)
// ---------------------------------------------------------------------------
constexpr int NB  = 4;              // batch
constexpr int SEQ = 2048;           // sequence length
constexpr int MD  = 512;            // model dim
constexpr int HH  = 8;              // heads
constexpr int DD  = 64;             // head dim
constexpr int FF  = 2048;           // ffn dim
constexpr int T   = NB * SEQ;       // 8192 tokens
constexpr int QKVC = 3 * HH * DD;   // 1536 fused qkv output cols

typedef __attribute__((ext_vector_type(16))) __bf16 v16bf;
typedef __attribute__((ext_vector_type(8)))  float  v8f;

union ABFrag {
    v16bf    v;
    uint32_t u[8];
};

// f32 -> bf16 round-to-nearest-even
__device__ __forceinline__ unsigned short f2bf(float f) {
    uint32_t u = __builtin_bit_cast(uint32_t, f);
    uint32_t r = (u + 0x7FFFu + ((u >> 16) & 1u)) >> 16;
    return (unsigned short)r;
}

// ---------------------------------------------------------------------------
// WMMA fragment loaders (wave32, 16x16x32 bf16)
// A (16x32, MxK): lane m = L%16, group g = L/16.
//   VGPR j: k = (j<4 ? 0 : 16) + 8*g + 2*(j&3), pair (k, k+1). Row-major src.
//   -> two contiguous 16B runs per lane; compiler emits global_load_b128 x2.
// B (32x16, KxN) from a transposed source Bt[n][k]: lane n = L%16, g = L/16.
//   VGPR j: k = (j<4 ? 0 : 16) + 8*g + 2*(j&3), pair (k, k+1) (same as A).
// C/D (16x16 f32): lane n = L%16, g = L/16; VGPR j holds row m = 8*g + j.
// ---------------------------------------------------------------------------
__device__ __forceinline__ void load_a_frag(ABFrag& fa, const unsigned short* src,
                                            int lda, int lane) {
    int m = lane & 15, g = lane >> 4;
    const unsigned short* row = src + (size_t)m * lda;
#pragma unroll
    for (int j = 0; j < 8; ++j) {
        int kk = ((j < 4) ? 0 : 16) + 8 * g + 2 * (j & 3);
        fa.u[j] = *(const uint32_t*)(row + kk);
    }
}

__device__ __forceinline__ void load_b_frag(ABFrag& fb, const unsigned short* bt,
                                            int ldb, int lane) {
    int n = lane & 15, g = lane >> 4;
    const unsigned short* row = bt + (size_t)n * ldb;
#pragma unroll
    for (int j = 0; j < 8; ++j) {
        int kk = ((j < 4) ? 0 : 16) + 8 * g + 2 * (j & 3);
        fb.u[j] = *(const uint32_t*)(row + kk);
    }
}

__device__ __forceinline__ v8f wmma_bf16(const ABFrag& a, const ABFrag& b, v8f c) {
    return __builtin_amdgcn_wmma_f32_16x16x32_bf16(
        /*neg_a=*/false, a.v, /*neg_b=*/false, b.v,
        /*c_mod=*/(short)0, c, /*reuse_a=*/false, /*reuse_b=*/false);
}

// ---------------------------------------------------------------------------
// 64x64 macro-tile GEMM core: one wave owns a 4x4 grid of 16x16 C tiles.
// Per k-step: 4 A-frags loaded once, then each B-frag is loaded and
// immediately consumed by its column of 4 WMMAs (minimizes live fragments:
// 128 C + 32 A + 8 B VGPRs). 16 WMMAs (256 KFLOP) per 8 KB of operands
// -> 32 FLOP/byte.
// A: [rows x Ka] row-major bf16 (row panel at t0); Bt: [cols x Ka] = B^T.
// ---------------------------------------------------------------------------
__device__ __forceinline__ void gemm_macro(const unsigned short* A,
                                           const unsigned short* Bt,
                                           int Ka, int lane, v8f c[4][4]) {
    const v8f zero = {0.f, 0.f, 0.f, 0.f, 0.f, 0.f, 0.f, 0.f};
#pragma unroll
    for (int mi = 0; mi < 4; ++mi)
#pragma unroll
        for (int ni = 0; ni < 4; ++ni)
            c[mi][ni] = zero;

    for (int k = 0; k < Ka; k += 32) {
        ABFrag fa[4];
#pragma unroll
        for (int mi = 0; mi < 4; ++mi)
            load_a_frag(fa[mi], A + (size_t)(mi * 16) * Ka + k, Ka, lane);
#pragma unroll
        for (int ni = 0; ni < 4; ++ni) {
            ABFrag fb;
            load_b_frag(fb, Bt + (size_t)(ni * 16) * Ka + k, Ka, lane);
#pragma unroll
            for (int mi = 0; mi < 4; ++mi)
                c[mi][ni] = wmma_bf16(fa[mi], fb, c[mi][ni]);
        }
    }
}

// ---------------------------------------------------------------------------
// Conversion / packing kernels
// ---------------------------------------------------------------------------
__global__ __launch_bounds__(256)
void cvt_bf16_kernel(const float* __restrict__ src,
                     unsigned short* __restrict__ dst, int n) {
    int i = blockIdx.x * blockDim.x + threadIdx.x;
    if (i < n) dst[i] = f2bf(src[i]);
}

// dst[c*R + r] = src[r*C + c]  (f32 -> bf16 transpose)
__global__ __launch_bounds__(256)
void transpose_cvt_kernel(const float* __restrict__ src,
                          unsigned short* __restrict__ dst,
                          int R, int C) {
    int i = blockIdx.x * blockDim.x + threadIdx.x;
    if (i >= R * C) return;
    int r = i / C, c = i % C;
    dst[(size_t)c * R + r] = f2bf(src[(size_t)r * C + c]);
}

// qkvT[c][m] = W{p}[h][m][d]  with c = p*512 + h*64 + d; also fused bias.
__global__ __launch_bounds__(256)
void pack_qkv_kernel(const float* __restrict__ Wq, const float* __restrict__ Wk,
                     const float* __restrict__ Wv, const float* __restrict__ bq,
                     const float* __restrict__ bk, const float* __restrict__ bv,
                     unsigned short* __restrict__ qkvT,
                     float* __restrict__ qkvBias) {
    int i = blockIdx.x * blockDim.x + threadIdx.x;
    if (i < QKVC * MD) {
        int c = i / MD, m = i % MD;
        int p = c >> 9, rem = c & 511, h = rem >> 6, d = rem & 63;
        const float* W = (p == 0) ? Wq : (p == 1) ? Wk : Wv;
        qkvT[(size_t)c * MD + m] = f2bf(W[((size_t)h * MD + m) * DD + d]);
    }
    if (i < QKVC) {
        int p = i >> 9, rem = i & 511;
        const float* B = (p == 0) ? bq : (p == 1) ? bk : bv;
        qkvBias[i] = B[rem];
    }
}

// ---------------------------------------------------------------------------
// Fused QKV projection: [8192 x 512] x [512 x 1536] -> scatter into
// Q[h][nb][q][d], K[h][nb][q][d], Vt[h][nb][d][q]  (all bf16).
// One wave = 64x64 macro-tile.
// ---------------------------------------------------------------------------
__global__ __launch_bounds__(256)
void gemm_qkv_kernel(const unsigned short* __restrict__ xb,
                     const unsigned short* __restrict__ qkvT,
                     const float* __restrict__ qkvBias,
                     unsigned short* __restrict__ Q,
                     unsigned short* __restrict__ Kh,
                     unsigned short* __restrict__ Vt) {
    int wid  = (blockIdx.x * blockDim.x + threadIdx.x) >> 5;
    int lane = threadIdx.x & 31;
    constexpr int NTC = QKVC / 64;   // 24
    int tc = wid % NTC, tr = wid / NTC;
    int t0 = tr * 64, c0 = tc * 64;

    v8f acc[4][4];
    gemm_macro(xb + (size_t)t0 * MD, qkvT + (size_t)c0 * MD, MD, lane, acc);

    int n = lane & 15, g = lane >> 4;
#pragma unroll
    for (int ni = 0; ni < 4; ++ni) {
        int c = c0 + ni * 16 + n;
        float bias = qkvBias[c];
        int p = c >> 9, h = (c >> 6) & 7, d = c & 63;
        size_t sliceQK = 0, sliceV = 0;
        if (p == 2) sliceV = (size_t)(h * NB) * DD * SEQ + (size_t)d * SEQ;
        else        sliceQK = (size_t)(h * NB) * SEQ * DD + d;
#pragma unroll
        for (int mi = 0; mi < 4; ++mi) {
#pragma unroll
            for (int j = 0; j < 8; ++j) {
                int t = t0 + mi * 16 + 8 * g + j;
                int nb = t >> 11, q = t & (SEQ - 1);
                unsigned short bv = f2bf(acc[mi][ni][j] + bias);
                if (p == 0)
                    Q [sliceQK + (size_t)nb * SEQ * DD + (size_t)q * DD] = bv;
                else if (p == 1)
                    Kh[sliceQK + (size_t)nb * SEQ * DD + (size_t)q * DD] = bv;
                else
                    Vt[sliceV + (size_t)nb * DD * SEQ + q] = bv;
            }
        }
    }
}

// ---------------------------------------------------------------------------
// Flash attention: one wave = 16 query rows x full 2048 keys for one (h, nb).
// Streaming softmax, P tile re-staged through per-wave LDS for the PV WMMA.
// Waves of a block share (h, nb) -> K/V panels hit in WGP$.
// ---------------------------------------------------------------------------
__global__ __launch_bounds__(256)
void attention_kernel(const unsigned short* __restrict__ Q,
                      const unsigned short* __restrict__ Kh,
                      const unsigned short* __restrict__ Vt,
                      unsigned short* __restrict__ concat) {
    __shared__ unsigned short smem[8][16 * 32];   // 1 KB per wave
    int widb = threadIdx.x >> 5, lane = threadIdx.x & 31;
    int wid  = blockIdx.x * 8 + widb;
    int qt = wid & 127;                // 2048/16 query tiles
    int hn = wid >> 7;                 // (h, nb)
    int h = hn >> 2, nb = hn & 3;

    const unsigned short* Qs = Q  + (size_t)(h * NB + nb) * SEQ * DD;
    const unsigned short* Ks = Kh + (size_t)(h * NB + nb) * SEQ * DD;
    const unsigned short* Vs = Vt + (size_t)(h * NB + nb) * DD * SEQ;

    int q0 = qt * 16;
    int n = lane & 15, g = lane >> 4;

    ABFrag aq0, aq1;
    load_a_frag(aq0, Qs + (size_t)q0 * DD + 0,  DD, lane);
    load_a_frag(aq1, Qs + (size_t)q0 * DD + 32, DD, lane);

    v8f O0 = {0.f,0.f,0.f,0.f,0.f,0.f,0.f,0.f};
    v8f O1 = O0, O2 = O0, O3 = O0;
    float run_m[8], run_l[8];
#pragma unroll
    for (int j = 0; j < 8; ++j) { run_m[j] = -INFINITY; run_l[j] = 0.f; }

    for (int kc = 0; kc < SEQ; kc += 32) {
        ABFrag fb;
        v8f S0 = {0.f,0.f,0.f,0.f,0.f,0.f,0.f,0.f};
        v8f S1 = S0;
        // scores: Q(16x64) x K^T — B source is K row-major [key][d]
        load_b_frag(fb, Ks + (size_t)kc * DD + 0,         DD, lane); S0 = wmma_bf16(aq0, fb, S0);
        load_b_frag(fb, Ks + (size_t)kc * DD + 32,        DD, lane); S0 = wmma_bf16(aq1, fb, S0);
        load_b_frag(fb, Ks + (size_t)(kc + 16) * DD + 0,  DD, lane); S1 = wmma_bf16(aq0, fb, S1);
        load_b_frag(fb, Ks + (size_t)(kc + 16) * DD + 32, DD, lane); S1 = wmma_bf16(aq1, fb, S1);

        float P0[8], P1[8];
#pragma unroll
        for (int j = 0; j < 8; ++j) {
            float s0 = S0[j] * 0.125f;      // 1/sqrt(64)
            float s1 = S1[j] * 0.125f;
            float tm = fmaxf(s0, s1);
            tm = fmaxf(tm, __shfl_xor(tm, 1, 32));
            tm = fmaxf(tm, __shfl_xor(tm, 2, 32));
            tm = fmaxf(tm, __shfl_xor(tm, 4, 32));
            tm = fmaxf(tm, __shfl_xor(tm, 8, 32));
            float mnew  = fmaxf(run_m[j], tm);
            float alpha = __expf(run_m[j] - mnew);
            float p0 = __expf(s0 - mnew), p1 = __expf(s1 - mnew);
            float rs = p0 + p1;
            rs += __shfl_xor(rs, 1, 32);
            rs += __shfl_xor(rs, 2, 32);
            rs += __shfl_xor(rs, 4, 32);
            rs += __shfl_xor(rs, 8, 32);
            run_l[j] = run_l[j] * alpha + rs;
            run_m[j] = mnew;
            O0[j] *= alpha; O1[j] *= alpha; O2[j] *= alpha; O3[j] *= alpha;
            P0[j] = p0; P1[j] = p1;
        }
        // C-layout P tile -> row-major 16x32 in LDS -> A operand.
        // LDS is per-wave private; DS pipeline is in-order within a wave.
#pragma unroll
        for (int j = 0; j < 8; ++j) {
            smem[widb][(8 * g + j) * 32 + n]      = f2bf(P0[j]);
            smem[widb][(8 * g + j) * 32 + 16 + n] = f2bf(P1[j]);
        }
        ABFrag pa;
        load_a_frag(pa, smem[widb], 32, lane);

        ABFrag bv;   // B source is Vt row-major [d][key]
        load_b_frag(bv, Vs + (size_t)0  * SEQ + kc, SEQ, lane); O0 = wmma_bf16(pa, bv, O0);
        load_b_frag(bv, Vs + (size_t)16 * SEQ + kc, SEQ, lane); O1 = wmma_bf16(pa, bv, O1);
        load_b_frag(bv, Vs + (size_t)32 * SEQ + kc, SEQ, lane); O2 = wmma_bf16(pa, bv, O2);
        load_b_frag(bv, Vs + (size_t)48 * SEQ + kc, SEQ, lane); O3 = wmma_bf16(pa, bv, O3);
    }

    int tok0 = nb * SEQ + q0;
#pragma unroll
    for (int j = 0; j < 8; ++j) {
        float inv = 1.f / run_l[j];
        size_t row = (size_t)(tok0 + 8 * g + j) * MD + h * DD + n;
        concat[row + 0 ] = f2bf(O0[j] * inv);
        concat[row + 16] = f2bf(O1[j] * inv);
        concat[row + 32] = f2bf(O2[j] * inv);
        concat[row + 48] = f2bf(O3[j] * inv);
    }
}

// ---------------------------------------------------------------------------
// 64x64 macro-tile GEMM + bias + residual -> f32 (out-proj, FFN2)
// ---------------------------------------------------------------------------
__global__ __launch_bounds__(256)
void gemm_bias_res_f32_kernel(const unsigned short* __restrict__ A,
                              const unsigned short* __restrict__ Bt,
                              const float* __restrict__ bias,
                              const float* __restrict__ R,
                              float* __restrict__ out,
                              int Ka, int Ncols) {
    int wid  = (blockIdx.x * blockDim.x + threadIdx.x) >> 5;
    int lane = threadIdx.x & 31;
    int ntc = Ncols >> 6;
    int tc = wid % ntc, tr = wid / ntc;
    int t0 = tr << 6, c0 = tc << 6;

    v8f acc[4][4];
    gemm_macro(A + (size_t)t0 * Ka, Bt + (size_t)c0 * Ka, Ka, lane, acc);

    int n = lane & 15, g = lane >> 4;
#pragma unroll
    for (int ni = 0; ni < 4; ++ni) {
        int c = c0 + ni * 16 + n;
        float bb = bias[c];
#pragma unroll
        for (int mi = 0; mi < 4; ++mi) {
#pragma unroll
            for (int j = 0; j < 8; ++j) {
                size_t idx = (size_t)(t0 + mi * 16 + 8 * g + j) * Ncols + c;
                out[idx] = acc[mi][ni][j] + bb + R[idx];
            }
        }
    }
}

// 64x64 macro-tile GEMM + bias + ReLU -> bf16 (FFN1)
__global__ __launch_bounds__(256)
void gemm_bias_relu_bf16_kernel(const unsigned short* __restrict__ A,
                                const unsigned short* __restrict__ Bt,
                                const float* __restrict__ bias,
                                unsigned short* __restrict__ out,
                                int Ka, int Ncols) {
    int wid  = (blockIdx.x * blockDim.x + threadIdx.x) >> 5;
    int lane = threadIdx.x & 31;
    int ntc = Ncols >> 6;
    int tc = wid % ntc, tr = wid / ntc;
    int t0 = tr << 6, c0 = tc << 6;

    v8f acc[4][4];
    gemm_macro(A + (size_t)t0 * Ka, Bt + (size_t)c0 * Ka, Ka, lane, acc);

    int n = lane & 15, g = lane >> 4;
#pragma unroll
    for (int ni = 0; ni < 4; ++ni) {
        int c = c0 + ni * 16 + n;
        float bb = bias[c];
#pragma unroll
        for (int mi = 0; mi < 4; ++mi) {
#pragma unroll
            for (int j = 0; j < 8; ++j) {
                size_t idx = (size_t)(t0 + mi * 16 + 8 * g + j) * Ncols + c;
                out[idx] = f2bf(fmaxf(acc[mi][ni][j] + bb, 0.f));
            }
        }
    }
}

// ---------------------------------------------------------------------------
// LayerNorm over M=512 (matches reference: std = sqrt(mean(zc^2)); eps on std)
// one block (256 thr) per token; optional bf16 copy of the output
// ---------------------------------------------------------------------------
__global__ __launch_bounds__(256)
void layernorm_kernel(const float* __restrict__ y,
                      const float* __restrict__ gamma,
                      const float* __restrict__ beta,
                      float* __restrict__ outf,
                      unsigned short* __restrict__ outb) {
    __shared__ float red[256];
    int t = blockIdx.x, tid = threadIdx.x;
    const float* row = y + (size_t)t * MD;
    float a = row[tid], b = row[tid + 256];

    red[tid] = a + b;
    __syncthreads();
    for (int o = 128; o > 0; o >>= 1) {
        if (tid < o) red[tid] += red[tid + o];
        __syncthreads();
    }
    float mean = red[0] * (1.f / MD);
    __syncthreads();

    float za = a - mean, zb = b - mean;
    red[tid] = za * za + zb * zb;
    __syncthreads();
    for (int o = 128; o > 0; o >>= 1) {
        if (tid < o) red[tid] += red[tid + o];
        __syncthreads();
    }
    float stdv = sqrtf(red[0] * (1.f / MD));
    float inv = 1.f / (stdv + 1e-10f);

    float ra = gamma[tid]       * (za * inv) + beta[tid];
    float rb = gamma[tid + 256] * (zb * inv) + beta[tid + 256];
    size_t base = (size_t)t * MD;
    if (outf) { outf[base + tid] = ra; outf[base + tid + 256] = rb; }
    if (outb) { outb[base + tid] = f2bf(ra); outb[base + tid + 256] = f2bf(rb); }
}

// ---------------------------------------------------------------------------
// Launch
// ---------------------------------------------------------------------------
extern "C" void kernel_launch(void* const* d_in, const int* in_sizes, int n_in,
                              void* d_out, int out_size, void* d_ws, size_t ws_size,
                              hipStream_t stream) {
    const float* x      = (const float*)d_in[0];
    const float* Wq     = (const float*)d_in[1];
    const float* bq     = (const float*)d_in[2];
    const float* Wk     = (const float*)d_in[3];
    const float* bk     = (const float*)d_in[4];
    const float* Wv     = (const float*)d_in[5];
    const float* bv     = (const float*)d_in[6];
    const float* Wp     = (const float*)d_in[7];
    const float* bp     = (const float*)d_in[8];
    const float* gamma1 = (const float*)d_in[9];
    const float* beta1  = (const float*)d_in[10];
    const float* W1     = (const float*)d_in[11];
    const float* b1     = (const float*)d_in[12];
    const float* W2     = (const float*)d_in[13];
    const float* b2     = (const float*)d_in[14];
    const float* gamma2 = (const float*)d_in[15];
    const float* beta2  = (const float*)d_in[16];
    float* out = (float*)d_out;

    char* ws = (char*)d_ws;
    size_t o = 0;
    auto alloc = [&](size_t bytes) -> char* {
        char* p = ws + o;
        o += (bytes + 255) & ~(size_t)255;
        return p;
    };
    unsigned short* xb      = (unsigned short*)alloc((size_t)T * MD * 2);
    unsigned short* qkvT    = (unsigned short*)alloc((size_t)QKVC * MD * 2);
    float*          qkvBias = (float*)         alloc((size_t)QKVC * 4);
    unsigned short* WpT     = (unsigned short*)alloc((size_t)MD * MD * 2);
    unsigned short* W1T     = (unsigned short*)alloc((size_t)FF * MD * 2);
    unsigned short* W2T     = (unsigned short*)alloc((size_t)MD * FF * 2);
    unsigned short* Qb      = (unsigned short*)alloc((size_t)HH * NB * SEQ * DD * 2);
    unsigned short* Kb      = (unsigned short*)alloc((size_t)HH * NB * SEQ * DD * 2);
    unsigned short* Vtb     = (unsigned short*)alloc((size_t)HH * NB * DD * SEQ * 2);
    unsigned short* concat  = (unsigned short*)alloc((size_t)T * MD * 2);
    float*          ybuf    = (float*)         alloc((size_t)T * MD * 4);   // reused for FFN2 out
    float*          x1f     = (float*)         alloc((size_t)T * MD * 4);
    unsigned short* x1b     = (unsigned short*)alloc((size_t)T * MD * 2);
    unsigned short* hmid    = (unsigned short*)alloc((size_t)T * FF * 2);

    const int BT = 256;

    // 1) activations + weights -> bf16 (weights transposed so B-operand
    //    fragment loads are contiguous dwords)
    cvt_bf16_kernel<<<(T * MD) / BT, BT, 0, stream>>>(x, xb, T * MD);
    pack_qkv_kernel<<<(QKVC * MD) / BT, BT, 0, stream>>>(Wq, Wk, Wv, bq, bk, bv, qkvT, qkvBias);
    transpose_cvt_kernel<<<(MD * MD) / BT, BT, 0, stream>>>(Wp, WpT, MD, MD);   // [HD][M]->[M][HD]
    transpose_cvt_kernel<<<(MD * FF) / BT, BT, 0, stream>>>(W1, W1T, MD, FF);   // [M][F]->[F][M]
    transpose_cvt_kernel<<<(FF * MD) / BT, BT, 0, stream>>>(W2, W2T, FF, MD);   // [F][M]->[M][F]

    // 2) fused QKV projection, 64x64 macro-tiles
    {
        int waves = (T / 64) * (QKVC / 64);          // 128*24 = 3072
        gemm_qkv_kernel<<<waves / 8, BT, 0, stream>>>(xb, qkvT, qkvBias, Qb, Kb, Vtb);
    }

    // 3) flash attention
    {
        int waves = (SEQ / 16) * HH * NB;            // 4096
        attention_kernel<<<waves / 8, BT, 0, stream>>>(Qb, Kb, Vtb, concat);
    }

    // 4) output projection + residual(x) -> y ; LN1 -> x1 (f32 + bf16)
    {
        int waves = (T / 64) * (MD / 64);            // 128*8 = 1024
        gemm_bias_res_f32_kernel<<<waves / 8, BT, 0, stream>>>(concat, WpT, bp, x, ybuf, MD, MD);
    }
    layernorm_kernel<<<T, BT, 0, stream>>>(ybuf, gamma1, beta1, x1f, x1b);

    // 5) FFN
    {
        int waves = (T / 64) * (FF / 64);            // 128*32 = 4096
        gemm_bias_relu_bf16_kernel<<<waves / 8, BT, 0, stream>>>(x1b, W1T, b1, hmid, MD, FF);
    }
    {
        int waves = (T / 64) * (MD / 64);            // 1024
        gemm_bias_res_f32_kernel<<<waves / 8, BT, 0, stream>>>(hmid, W2T, b2, x1f, ybuf, FF, MD);
    }

    // 6) LN2 -> final f32 output
    layernorm_kernel<<<T, BT, 0, stream>>>(ybuf, gamma2, beta2, out, nullptr);
}
